// AlignViTBlock_55052890800921
// MI455X (gfx1250) — matrix operand order repaired and checked by reading, not compiled, not run on hardware
//
#include <hip/hip_runtime.h>

typedef __attribute__((ext_vector_type(8)))  _Float16 v8h;
typedef __attribute__((ext_vector_type(16))) _Float16 v16h;
typedef __attribute__((ext_vector_type(8)))  float    v8f;

#define TOK   8192   // B*H*W = 2*64*64
#define CDIM  128
#define NHEAD 4
#define DHD   32
#define HDIM  64
#define WDIM  64
#define KWIN  7

#define GF_RELU   1
#define GF_QSCALE 2
#define GF_TROUT  4

// ---------------- all five weight matrices -> one contiguous f16 region ----------------
// layout: [0,49152) qkv_w | [49152,65536) proj_w | [65536,81920) w1 | [81920,98304) w2 | [98304,114688) w3
__global__ void cvt_all(const float* __restrict__ qkvw, const float* __restrict__ projw,
                        const float* __restrict__ w1, const float* __restrict__ w2,
                        const float* __restrict__ w3, _Float16* __restrict__ dst) {
  int i = blockIdx.x * 256 + threadIdx.x;   // 114688 total; boundaries are multiples of 256
  const float* s; int off;
  if (i < 49152)      { s = qkvw;  off = i; }
  else if (i < 65536) { s = projw; off = i - 49152; }
  else if (i < 81920) { s = w1;    off = i - 65536; }
  else if (i < 98304) { s = w2;    off = i - 81920; }
  else                { s = w3;    off = i - 98304; }
  dst[i] = (_Float16)s[off];
}

// ---------------- LayerNorm over C=128, gather from BCHW ----------------
// one wave per token; writes fp32 residual copy (token-major) + f16 LN output
__global__ void ln_first(const float* __restrict__ x, const float* __restrict__ g,
                         const float* __restrict__ b, float* __restrict__ xr,
                         _Float16* __restrict__ o16) {
  int lane = threadIdx.x & 31;
  int t = blockIdx.x * 8 + (threadIdx.x >> 5);
  int bb = t >> 12, p = t & 4095;
  float v[4];
  float s = 0.f, s2 = 0.f;
#pragma unroll
  for (int i = 0; i < 4; i++) {
    int c = lane + 32 * i;
    float vv = x[(size_t)(bb * CDIM + c) * 4096 + p];
    v[i] = vv; s += vv; s2 += vv * vv;
  }
#pragma unroll
  for (int off = 16; off; off >>= 1) { s += __shfl_xor(s, off); s2 += __shfl_xor(s2, off); }
  float mu = s * (1.f / CDIM);
  float rs = rsqrtf(s2 * (1.f / CDIM) - mu * mu + 1e-5f);
#pragma unroll
  for (int i = 0; i < 4; i++) {
    int c = lane + 32 * i;
    xr[(size_t)t * CDIM + c] = v[i];
    o16[(size_t)t * CDIM + c] = (_Float16)((v[i] - mu) * rs * g[c] + b[c]);
  }
}

// ---------------- WMMA GEMM, 16(M) x 64(N) per wave, K=128 ----------------
// out[M=TOK,N] = A(f16 MxK) @ W(f16 NxK)^T + bias; A fragment reused across 4 N-tiles.
__global__ void gemm_nt64(const _Float16* __restrict__ A, const _Float16* __restrict__ W,
                          const float* __restrict__ bias, const float* __restrict__ resid,
                          float* __restrict__ out, int N, int flags) {
  const int K = 128;
  int wave = blockIdx.x * (blockDim.x >> 5) + (threadIdx.x >> 5);
  int tilesN = N >> 6;
  int tm  = (wave / tilesN) << 4;
  int tn0 = (wave % tilesN) << 6;
  int l = threadIdx.x & 31;
  int r = l & 15, hi = l >> 4;
  v8f acc[4];
#pragma unroll
  for (int j = 0; j < 4; j++) acc[j] = (v8f){};
#pragma unroll
  for (int kb = 0; kb < K / 32; ++kb) {
    // A fragment: row m = tm+r; lanes 0-15 K={0..7,16..23}, lanes 16-31 K={8..15,24..31}
    const _Float16* ap = A + (size_t)(tm + r) * K + kb * 32 + hi * 8;
    v8h a0 = *(const v8h*)ap;
    v8h a1 = *(const v8h*)(ap + 16);
    v16h af;
#pragma unroll
    for (int i = 0; i < 8; i++) { af[i] = a0[i]; af[i + 8] = a1[i]; }
#pragma unroll
    for (int j = 0; j < 4; j++) {
      // B fragment: col n = tn0+j*16+r (B[k][n]=W[n][k]); lanes 0-15 K={0..7,16..23}, 16-31 K={8..15,24..31}
      const _Float16* bp = W + (size_t)(tn0 + j * 16 + r) * K + kb * 32 + hi * 8;
      v8h b0 = *(const v8h*)bp;
      v8h b1 = *(const v8h*)(bp + 16);
      v16h bf;
#pragma unroll
      for (int i = 0; i < 8; i++) { bf[i] = b0[i]; bf[i + 8] = b1[i]; }
      acc[j] = __builtin_amdgcn_wmma_f32_16x16x32_f16(false, af, false, bf, (short)0, acc[j], false, false);
    }
  }
#pragma unroll
  for (int j = 0; j < 4; j++) {
    int n = tn0 + j * 16 + r;
    float bn = bias[n];
#pragma unroll
    for (int r2 = 0; r2 < 8; r2++) {
      int m = tm + hi * 8 + r2;                 // C/D layout: lanes<16 rows 0..7, lanes>=16 rows 8..15
      float val = acc[j][r2] + bn;
      if (flags & GF_QSCALE) { if (n < 128) val *= 0.17677669529663687f; }  // DH^-0.5
      if (flags & GF_RELU) val = fmaxf(val, 0.f);
      if (resid) val += resid[(size_t)m * CDIM + n];
      if (flags & GF_TROUT) {
        int bb = m >> 12, p = m & 4095;         // BHWC -> BCHW
        out[(size_t)(bb * CDIM + n) * 4096 + p] = val;
      } else {
        out[(size_t)m * N + n] = val;
      }
    }
  }
}

// ---------------- fused WMMA GEMM (16 rows x all 128 cols per wave) + epilogue LayerNorm ----
// val = A@W^T + bias [+ resid] [ReLU]; optionally write val f32; always write LN(val) f16.
// Safe to run in place (out16 == A): each wave reads only the 16-row band it writes.
__global__ void gemm_row_ln(const _Float16* __restrict__ A, const _Float16* __restrict__ W,
                            const float* __restrict__ bias, const float* __restrict__ resid,
                            const float* __restrict__ g, const float* __restrict__ be,
                            float* __restrict__ out32, _Float16* __restrict__ out16, int flags) {
  const int K = 128;
  int wave = blockIdx.x * (blockDim.x >> 5) + (threadIdx.x >> 5);
  int tm = wave << 4;
  int l = threadIdx.x & 31;
  int r = l & 15, hi = l >> 4;
  v8f acc[8];
#pragma unroll
  for (int j = 0; j < 8; j++) acc[j] = (v8f){};
#pragma unroll
  for (int kb = 0; kb < K / 32; ++kb) {
    const _Float16* ap = A + (size_t)(tm + r) * K + kb * 32 + hi * 8;
    v8h a0 = *(const v8h*)ap;
    v8h a1 = *(const v8h*)(ap + 16);
    v16h af;
#pragma unroll
    for (int i = 0; i < 8; i++) { af[i] = a0[i]; af[i + 8] = a1[i]; }
#pragma unroll
    for (int j = 0; j < 8; j++) {
      const _Float16* bp = W + (size_t)(j * 16 + r) * K + kb * 32 + hi * 8;
      v8h b0 = *(const v8h*)bp;
      v8h b1 = *(const v8h*)(bp + 16);
      v16h bf;
#pragma unroll
      for (int i = 0; i < 8; i++) { bf[i] = b0[i]; bf[i + 8] = b1[i]; }
      acc[j] = __builtin_amdgcn_wmma_f32_16x16x32_f16(false, af, false, bf, (short)0, acc[j], false, false);
    }
  }
  // epilogue: bias / resid / relu (+ optional f32 store of pre-LN value)
#pragma unroll
  for (int j = 0; j < 8; j++) {
    int n = j * 16 + r;
    float bn = bias[n];
#pragma unroll
    for (int r2 = 0; r2 < 8; r2++) {
      int m = tm + hi * 8 + r2;
      float v = acc[j][r2] + bn;
      if (resid) v += resid[(size_t)m * CDIM + n];
      if (flags & GF_RELU) v = fmaxf(v, 0.f);
      acc[j][r2] = v;
      if (out32) out32[(size_t)m * CDIM + n] = v;
    }
  }
  // per-row LN stats: row m's 128 values are 8 regs x 16 lanes of one half-wave
  float mu[8], rs[8];
#pragma unroll
  for (int r2 = 0; r2 < 8; r2++) {
    float s = 0.f, s2 = 0.f;
#pragma unroll
    for (int j = 0; j < 8; j++) { float v = acc[j][r2]; s += v; s2 += v * v; }
#pragma unroll
    for (int off = 8; off; off >>= 1) { s += __shfl_xor(s, off); s2 += __shfl_xor(s2, off); }
    float m_ = s * (1.f / CDIM);
    mu[r2] = m_;
    rs[r2] = rsqrtf(s2 * (1.f / CDIM) - m_ * m_ + 1e-5f);
  }
#pragma unroll
  for (int j = 0; j < 8; j++) {
    int n = j * 16 + r;
    float gn = g[n], b2 = be[n];
#pragma unroll
    for (int r2 = 0; r2 < 8; r2++) {
      int m = tm + hi * 8 + r2;
      out16[(size_t)m * CDIM + n] = (_Float16)((acc[j][r2] - mu[r2]) * rs[r2] * gn + b2);
    }
  }
}

// ---------------- 7x7 neighborhood attention, one wave per (token, head) ----------------
// qkv: (TOK, 384) fp32, q pre-scaled; rpb: (4,13,13); out: f16 (TOK,128) head-major cols
__global__ void na2d(const float* __restrict__ qkv, const float* __restrict__ rpb,
                     _Float16* __restrict__ o16) {
  int lane = threadIdx.x & 31;
  int gid = blockIdx.x * 8 + (threadIdx.x >> 5);
  int t = gid >> 2;
  int head = gid & 3;
  int bb = t >> 12, p = t & 4095, h = p >> 6, w = p & 63;
  int sh = h - (KWIN / 2); sh = sh < 0 ? 0 : (sh > HDIM - KWIN ? HDIM - KWIN : sh);
  int sw = w - (KWIN / 2); sw = sw < 0 ? 0 : (sw > WDIM - KWIN ? WDIM - KWIN : sw);
  float q = qkv[(size_t)t * 384 + head * DHD + lane];
  const float* rp = rpb + head * 169 + (sh - h + KWIN - 1) * 13 + (sw - w + KWIN - 1);
  float m = -1e30f, den = 0.f, acc = 0.f;
  for (int i = 0; i < KWIN; i++) {
    for (int j = 0; j < KWIN; j++) {
      int nt = (bb << 12) + ((sh + i) << 6) + (sw + j);
      const float* nb = qkv + (size_t)nt * 384 + head * DHD + lane;
      float s = q * nb[128];                               // q . k partial
#pragma unroll
      for (int off = 16; off; off >>= 1) s += __shfl_xor(s, off);
      s += rp[i * 13 + j];
      float mn = fmaxf(m, s);
      float sc = __expf(m - mn);
      float pe = __expf(s - mn);
      den = den * sc + pe;
      acc = acc * sc + pe * nb[256];                       // online softmax * v
      m = mn;
    }
  }
  o16[(size_t)t * CDIM + head * DHD + lane] = (_Float16)(acc / den);
}

extern "C" void kernel_launch(void* const* d_in, const int* in_sizes, int n_in,
                              void* d_out, int out_size, void* d_ws, size_t ws_size,
                              hipStream_t stream) {
  (void)in_sizes; (void)n_in; (void)out_size; (void)ws_size;
  const float* x      = (const float*)d_in[0];
  const float* g1     = (const float*)d_in[1];
  const float* be1    = (const float*)d_in[2];
  const float* g2     = (const float*)d_in[3];
  const float* be2    = (const float*)d_in[4];
  const float* g3     = (const float*)d_in[5];
  const float* be3    = (const float*)d_in[6];
  const float* g4     = (const float*)d_in[7];
  const float* be4    = (const float*)d_in[8];
  const float* qkv_w  = (const float*)d_in[9];
  const float* qkv_b  = (const float*)d_in[10];
  const float* rpb    = (const float*)d_in[11];
  const float* proj_w = (const float*)d_in[12];
  const float* proj_b = (const float*)d_in[13];
  const float* w1     = (const float*)d_in[14];
  const float* bl1    = (const float*)d_in[15];
  const float* w2     = (const float*)d_in[16];
  const float* bl2    = (const float*)d_in[17];
  const float* w3     = (const float*)d_in[18];
  const float* bl3    = (const float*)d_in[19];
  float* out = (float*)d_out;

  char* pws = (char*)d_ws;
  auto carve = [&](size_t bytes) -> void* {
    void* r = (void*)pws; pws += (bytes + 255) & ~(size_t)255; return r;
  };
  _Float16* w16base = (_Float16*)carve((size_t)114688 * 2);       // all weights f16
  _Float16* qkvw16  = w16base;
  _Float16* projw16 = w16base + 49152;
  _Float16* w116    = w16base + 65536;
  _Float16* w216    = w16base + 81920;
  _Float16* w316    = w16base + 98304;
  _Float16* a16     = (_Float16*)carve((size_t)TOK * CDIM * 2);   // f16 activations (reused, in-place)
  float*    xr      = (float*)carve((size_t)TOK * CDIM * 4);      // residual 1
  float*    x1      = (float*)carve((size_t)TOK * CDIM * 4);      // residual 2
  float*    qkvbuf  = (float*)carve((size_t)TOK * 384 * 4);       // qkv activations

  // 1) all weight conversions in one launch
  cvt_all<<<114688 / 256, 256, 0, stream>>>(qkv_w, proj_w, w1, w2, w3, w16base);
  // 2) LN1 (BCHW gather) -> residual xr + f16 a16
  ln_first<<<TOK / 8, 256, 0, stream>>>(x, g1, be1, xr, a16);
  // 3) QKV GEMM (8192x384x128), q pre-scaled
  gemm_nt64<<<(TOK / 16) * (384 / 64) / 8, 256, 0, stream>>>(a16, qkvw16, qkv_b, nullptr, qkvbuf, 384, GF_QSCALE);
  // 4) neighborhood attention -> a16 (f16)
  na2d<<<TOK * NHEAD / 8, 256, 0, stream>>>(qkvbuf, rpb, a16);
  // 5) proj GEMM + LN2 fused, in place on a16
  gemm_row_ln<<<(TOK / 16) / 8, 256, 0, stream>>>(a16, projw16, proj_b, nullptr, g2, be2, nullptr, a16, 0);
  // 6) w1 GEMM + residual(xr) -> x1 (f32) and LN3 fused -> a16
  gemm_row_ln<<<(TOK / 16) / 8, 256, 0, stream>>>(a16, w116, bl1, xr, g3, be3, x1, a16, 0);
  // 7) w2 GEMM + ReLU + LN4 fused, in place on a16
  gemm_row_ln<<<(TOK / 16) / 8, 256, 0, stream>>>(a16, w216, bl2, nullptr, g4, be4, nullptr, a16, GF_RELU);
  // 8) w3 GEMM + residual(x1) + transposed BCHW store to d_out
  gemm_nt64<<<(TOK / 16) * (CDIM / 64) / 8, 256, 0, stream>>>(a16, w316, bl3, x1, out, CDIM, GF_TROUT);
}
